// MPNEncoder_51634096832942
// MI455X (gfx1250) — compile-verified, matrix operand order checked by reading, not run on hardware
//
#include <hip/hip_runtime.h>
#include <hip/hip_bf16.h>
#include <stdint.h>

// ---------------------------------------------------------------------------
// D-MPNN (chemprop) encoder for gfx1250 (MI455X), bf16 WMMA pipeline.
// messages kept in bf16 so the 200000x320 message array (128MB) fits the
// 192MB L2 -> gather passes run at L2 bandwidth; GEMMs on v_wmma bf16 with
// double-buffered LDS staging via GLOBAL_LOAD_ASYNC_TO_LDS (ASYNCcnt).
// ---------------------------------------------------------------------------

typedef __attribute__((ext_vector_type(16))) __bf16 v16bf;
typedef __attribute__((ext_vector_type(8)))  __bf16 v8bf;
typedef __attribute__((ext_vector_type(8)))  float  v8f;
typedef __attribute__((ext_vector_type(4)))  int    v4i;

#ifndef __has_builtin
#define __has_builtin(x) 0
#endif

#if defined(__AMDGCN__) && \
    __has_builtin(__builtin_amdgcn_global_load_async_to_lds_b128) && \
    __has_builtin(__builtin_amdgcn_s_wait_asynccnt)
#define USE_ASYNC_LDS 1
#else
#define USE_ASYNC_LDS 0
#endif

#define AF    133
#define BFD   147
#define HDIM  300
#define MAXNB 6
#define HP    320   // H padded to multiple of 32 (WMMA K-step)
#define KBI   160   // bond fdim padded
#define KAO   448   // (AF + H) padded
#define MBLK  128
#define NBLK  64
#define KSTEP 32

// ----- bf16 <-> f32 helpers (bit-exact RNE truncation, no cast reliance) ---
__device__ __forceinline__ __bf16 f2bf(float x) {
  union { float f; uint32_t u; } c; c.f = x;
  uint32_t r = c.u + 0x7FFFu + ((c.u >> 16) & 1u);
  uint16_t h = (uint16_t)(r >> 16);
  __bf16 o; __builtin_memcpy(&o, &h, 2); return o;
}
__device__ __forceinline__ float bf2f(__bf16 x) {
  uint16_t h; __builtin_memcpy(&h, &x, 2);
  union { uint32_t u; float f; } c; c.u = ((uint32_t)h) << 16;
  return c.f;
}

// 16B memory->LDS mover: async on gfx1250 when available, else via VGPRs.
__device__ __forceinline__ void move16_to_lds(const __bf16* gp, __bf16* lp) {
#if USE_ASYNC_LDS
  // Builtin signature (from compiler diagnostic): param0 = AS1 v4i*, i.e.
  // (global int4* src, local int4* dst, imm offset, imm cpol).
  v4i* gg = (v4i*)(void*)gp;
  v4i* ll = (v4i*)(void*)lp;
  __builtin_amdgcn_global_load_async_to_lds_b128(
      (__attribute__((address_space(1))) v4i*)gg,
      (__attribute__((address_space(3))) v4i*)ll, 0, 0);
#else
  *(v8bf*)lp = *(const v8bf*)gp;
#endif
}

// ---------------------------------------------------------------------------
// WMMA GEMM: C[M][outld] = A[M][lda](bf16) * Bt^T   (Bt is [N][ldb] bf16)
// MODE 0: out0 = C (bf16), out1 = relu(C) (bf16)          [W_i]
// MODE 1: out1 = relu(C + inpB) (bf16)                    [W_h]
// MODE 2: outF = relu(C + bias[col]) (f32)                [W_o]
// Block: 256 thr = 8 waves; block tile 128x64; wave tile 32x32 (2x2 WMMA);
// LDS double-buffered (24 KB), 3 async-LDS ops per wave per stage.
// ---------------------------------------------------------------------------
template<int MODE>
__global__ __launch_bounds__(256)
void gemm_wmma_kernel(const __bf16* __restrict__ A, int lda, int M, int ksteps,
                      const __bf16* __restrict__ Bt, int ldb,
                      const __bf16* __restrict__ inpB,
                      const float*  __restrict__ bias,
                      __bf16* __restrict__ out0,
                      __bf16* __restrict__ out1,
                      float*  __restrict__ outF,
                      int outld)
{
  __shared__ __align__(16) __bf16 sA[2 * MBLK * KSTEP];  // 16 KB
  __shared__ __align__(16) __bf16 sB[2 * NBLK * KSTEP];  //  8 KB

  const int tid  = threadIdx.x;
  const int wave = tid >> 5;
  const int lane = tid & 31;
  const int half = lane >> 4;   // selects K-offset 8 per ISA A/B layout
  const int l16  = lane & 15;

  const int mblock = blockIdx.y * MBLK;
  const int nblock = blockIdx.x * NBLK;
  const int wm = (wave & 3) * 32;   // 4 waves along M
  const int wn = (wave >> 2) * 32;  // 2 waves along N

  // Per-thread staging coordinates (branch-free; A rows clamped: rows >= M
  // are never stored by the epilogue, so stale/duplicated data is harmless).
  const int arow0 = tid >> 2;             // A chunk 0 row (0..63)
  const int arow1 = (tid + 256) >> 2;     // A chunk 1 row (64..127)
  const int acc_  = (tid & 3) * 8;        // chunk column offset (bf16 elems)
  int ag0 = mblock + arow0; if (ag0 >= M) ag0 = M - 1;
  int ag1 = mblock + arow1; if (ag1 >= M) ag1 = M - 1;
  const int brow = tid >> 2;              // B row (0..63)

  // Stage tile (A:128x32, B:64x32) into LDS buffer `buf` at k-offset k0.
  auto stage = [&](int buf, int k0) {
    __bf16* bA = sA + buf * (MBLK * KSTEP);
    __bf16* bB = sB + buf * (NBLK * KSTEP);
    move16_to_lds(A + (size_t)ag0 * lda + k0 + acc_, bA + arow0 * KSTEP + acc_);
    move16_to_lds(A + (size_t)ag1 * lda + k0 + acc_, bA + arow1 * KSTEP + acc_);
    move16_to_lds(Bt + (size_t)(nblock + brow) * ldb + k0 + acc_,
                  bB + brow * KSTEP + acc_);
  };

  v8f acc[2][2];
#pragma unroll
  for (int i = 0; i < 2; ++i)
#pragma unroll
    for (int j = 0; j < 2; ++j)
#pragma unroll
      for (int r = 0; r < 8; ++r) acc[i][j][r] = 0.0f;

  stage(0, 0);  // prologue

  for (int ks = 0; ks < ksteps; ++ks) {
    const int buf = ks & 1;
    if (ks + 1 < ksteps) stage(buf ^ 1, (ks + 1) * KSTEP);
#if USE_ASYNC_LDS
    // Async ops retire in order: <=3 outstanding means current stage's 3
    // ops have landed in LDS (next stage's 3 may still be in flight).
    if (ks + 1 < ksteps) __builtin_amdgcn_s_wait_asynccnt(3);
    else                 __builtin_amdgcn_s_wait_asynccnt(0);
#endif
    __syncthreads();

    const __bf16* bA = sA + buf * (MBLK * KSTEP);
    const __bf16* bB = sB + buf * (NBLK * KSTEP);

    // ---- fragments per ISA 16-bit 16x32 layout:
    // lanes 0-15: V0..3 = K[0..7], V4..7 = K[16..23]; lanes 16-31: +8 ----
    v16bf afrag[2], bfrag[2];
#pragma unroll
    for (int i = 0; i < 2; ++i) {
      const __bf16* pa = bA + (wm + i * 16 + l16) * KSTEP + half * 8;
      v8bf lo = *(const v8bf*)pa;
      v8bf hi = *(const v8bf*)(pa + 16);
#pragma unroll
      for (int e = 0; e < 8; ++e) { afrag[i][e] = lo[e]; afrag[i][8 + e] = hi[e]; }
    }
#pragma unroll
    for (int j = 0; j < 2; ++j) {
      const __bf16* pb = bB + (wn + j * 16 + l16) * KSTEP + half * 8;
      v8bf lo = *(const v8bf*)pb;
      v8bf hi = *(const v8bf*)(pb + 16);
#pragma unroll
      for (int e = 0; e < 8; ++e) { bfrag[j][e] = lo[e]; bfrag[j][8 + e] = hi[e]; }
    }

#pragma unroll
    for (int i = 0; i < 2; ++i)
#pragma unroll
      for (int j = 0; j < 2; ++j)
        acc[i][j] = __builtin_amdgcn_wmma_f32_16x16x32_bf16(
            false, afrag[i], false, bfrag[j], (short)0, acc[i][j], false, false);

    __syncthreads();  // consumers done before buffer is overwritten
  }

  // ---- epilogue (C/D layout: VGPR r -> M = r + half*8; N = l16) ----
#pragma unroll
  for (int i = 0; i < 2; ++i) {
#pragma unroll
    for (int j = 0; j < 2; ++j) {
      const int col = nblock + wn + j * 16 + l16;
#pragma unroll
      for (int r = 0; r < 8; ++r) {
        const int row = mblock + wm + i * 16 + half * 8 + r;
        if (row < M) {
          float v = acc[i][j][r];
          size_t o = (size_t)row * outld + col;
          if (MODE == 0) {
            out0[o] = f2bf(v);
            out1[o] = f2bf(fmaxf(v, 0.0f));
          } else if (MODE == 1) {
            float s = v + bf2f(inpB[o]);
            out1[o] = f2bf(fmaxf(s, 0.0f));
          } else {
            outF[o] = fmaxf(v + bias[col], 0.0f);
          }
        }
      }
    }
  }
}

// ---------------------------------------------------------------------------
// Elementwise / data-movement kernels
// ---------------------------------------------------------------------------
__global__ void cvt_pad_rows_kernel(const float* __restrict__ src, int M, int Ksrc,
                                    __bf16* __restrict__ dst, int Kdst) {
  int idx = blockIdx.x * blockDim.x + threadIdx.x;
  if (idx >= M * Kdst) return;
  int r = idx / Kdst, k = idx - r * Kdst;
  dst[idx] = (k < Ksrc) ? f2bf(src[(size_t)r * Ksrc + k]) : f2bf(0.0f);
}

// dst[n][k] = W[k][n] (bf16), padded to [Nrows=HP][Kp]
__global__ void transpose_pad_kernel(const float* __restrict__ W, int Kreal, int Kp,
                                     __bf16* __restrict__ dst) {
  int idx = blockIdx.x * blockDim.x + threadIdx.x;
  if (idx >= HP * Kp) return;
  int n = idx / Kp, k = idx - n * Kp;
  dst[idx] = (n < HDIM && k < Kreal) ? f2bf(W[(size_t)k * HDIM + n]) : f2bf(0.0f);
}

__global__ void pad_bias_kernel(const float* __restrict__ b, float* __restrict__ dst) {
  int c = blockIdx.x * blockDim.x + threadIdx.x;
  if (c < HP) dst[c] = (c < HDIM) ? b[c] : 0.0f;
}

// a_msg[a] = sum_j msg[a2b[a][j]]   (16B bf16 chunks, fp32 accumulate)
__global__ void aggregate_kernel(const __bf16* __restrict__ msg,
                                 const int* __restrict__ a2b, int na,
                                 __bf16* __restrict__ out) {
  int idx = blockIdx.x * blockDim.x + threadIdx.x;
  const int chunks = HP / 8;
  if (idx >= na * chunks) return;
  int a = idx / chunks, cc = (idx - a * chunks) * 8;
  float s[8];
#pragma unroll
  for (int e = 0; e < 8; ++e) s[e] = 0.0f;
  for (int j = 0; j < MAXNB; ++j) {
    int b = a2b[a * MAXNB + j];
    v8bf v = *(const v8bf*)(msg + (size_t)b * HP + cc);
#pragma unroll
    for (int e = 0; e < 8; ++e) s[e] += bf2f(v[e]);
  }
  v8bf o;
#pragma unroll
  for (int e = 0; e < 8; ++e) o[e] = f2bf(s[e]);
  *(v8bf*)(out + (size_t)a * HP + cc) = o;
}

// bnd[b] = a_msg[b2a[b]] - msg[b2revb[b]]
__global__ void bond_build_kernel(const __bf16* __restrict__ amsg,
                                  const __bf16* __restrict__ msg,
                                  const int* __restrict__ b2a,
                                  const int* __restrict__ b2revb, int nb,
                                  __bf16* __restrict__ out) {
  int idx = blockIdx.x * blockDim.x + threadIdx.x;
  const int chunks = HP / 8;
  if (idx >= nb * chunks) return;
  int b = idx / chunks, cc = (idx - b * chunks) * 8;
  int a = b2a[b], rb = b2revb[b];
  v8bf va = *(const v8bf*)(amsg + (size_t)a  * HP + cc);
  v8bf vm = *(const v8bf*)(msg  + (size_t)rb * HP + cc);
  v8bf o;
#pragma unroll
  for (int e = 0; e < 8; ++e) o[e] = f2bf(bf2f(va[e]) - bf2f(vm[e]));
  *(v8bf*)(out + (size_t)b * HP + cc) = o;
}

// a_input = concat(f_atoms, a_msg) padded to KAO
__global__ void build_ainput_kernel(const float* __restrict__ f_atoms,
                                    const __bf16* __restrict__ amsg, int na,
                                    __bf16* __restrict__ out) {
  int idx = blockIdx.x * blockDim.x + threadIdx.x;
  if (idx >= na * KAO) return;
  int a = idx / KAO, c = idx - a * KAO;
  __bf16 v;
  if (c < AF)             v = f2bf(f_atoms[(size_t)a * AF + c]);
  else if (c < AF + HDIM) v = amsg[(size_t)a * HP + (c - AF)];
  else                    v = f2bf(0.0f);
  out[idx] = v;
}

__global__ void zero_f32_kernel(float* __restrict__ p, int n) {
  int idx = blockIdx.x * blockDim.x + threadIdx.x;
  if (idx < n) p[idx] = 0.0f;
}

__global__ void count_kernel(const int* __restrict__ atom_mol, int na,
                             float* __restrict__ counts) {
  int idx = blockIdx.x * blockDim.x + threadIdx.x;
  if (idx < na) atomicAdd(&counts[atom_mol[idx]], 1.0f);
}

__global__ void scatter_hidden_kernel(const float* __restrict__ hid,
                                      const int* __restrict__ atom_mol, int na,
                                      float* __restrict__ sums) {
  int idx = blockIdx.x * blockDim.x + threadIdx.x;
  if (idx >= na * HDIM) return;
  int a = idx / HDIM, c = idx - a * HDIM;
  atomicAdd(&sums[(size_t)atom_mol[a] * HDIM + c], hid[(size_t)a * HP + c]);
}

__global__ void divide_kernel(float* __restrict__ out, const float* __restrict__ counts,
                              int nm) {
  int idx = blockIdx.x * blockDim.x + threadIdx.x;
  if (idx >= nm * HDIM) return;
  out[idx] = out[idx] / fmaxf(counts[idx / HDIM], 1.0f);
}

// ---------------------------------------------------------------------------
extern "C" void kernel_launch(void* const* d_in, const int* in_sizes, int n_in,
                              void* d_out, int out_size, void* d_ws, size_t ws_size,
                              hipStream_t stream) {
  const float* f_atoms  = (const float*)d_in[0];
  const float* f_bonds  = (const float*)d_in[1];
  const float* W_i      = (const float*)d_in[2];
  const float* W_h      = (const float*)d_in[3];
  const float* W_o      = (const float*)d_in[4];
  const float* b_o      = (const float*)d_in[5];
  const int*   a2b      = (const int*)d_in[6];
  const int*   b2a      = (const int*)d_in[7];
  const int*   b2revb   = (const int*)d_in[8];
  const int*   atom_mol = (const int*)d_in[9];

  const int na = in_sizes[0] / AF;
  const int nb = in_sizes[1] / BFD;
  const int nm = out_size / HDIM;

  char* ws = (char*)d_ws;
  size_t off = 0;
  auto carve = [&](size_t bytes) -> char* {
    char* p = ws + off;
    off = (off + bytes + 255) & ~(size_t)255;
    return p;
  };

  __bf16* wi_t  = (__bf16*)carve((size_t)HP * KBI * 2);
  __bf16* wh_t  = (__bf16*)carve((size_t)HP * HP  * 2);
  __bf16* wo_t  = (__bf16*)carve((size_t)HP * KAO * 2);
  float*  biasp = (float*) carve((size_t)HP * 4);
  float*  counts= (float*) carve((size_t)nm * 4);

  // Aliased big regions (phase-disjoint lifetimes):
  size_t szA = (size_t)nb * KBI * 2; { size_t s2 = (size_t)na * HP * 2; if (s2 > szA) szA = s2; }
  char* regA = carve(szA);                       // fb16  -> amsg
  char* regB = carve((size_t)nb * HP * 2);       // inp   -> hid (na*HP*4 == nb*HP*2)
  char* regC = carve((size_t)nb * HP * 2);       // msg
  size_t szD = (size_t)nb * HP * 2; { size_t s2 = (size_t)na * KAO * 2; if (s2 > szD) szD = s2; }
  char* regD = carve(szD);                       // bnd   -> ain
  (void)ws_size;

  __bf16* fb16 = (__bf16*)regA;  __bf16* amsg = (__bf16*)regA;
  __bf16* inp  = (__bf16*)regB;  float*  hid  = (float*) regB;
  __bf16* msg  = (__bf16*)regC;
  __bf16* bnd  = (__bf16*)regD;  __bf16* ain  = (__bf16*)regD;

  auto cdiv = [](long long a, long long b) { return (int)((a + b - 1) / b); };
  const int T = 256;

  // --- precision conversion / weight transposition ---
  cvt_pad_rows_kernel<<<cdiv((long long)nb * KBI, T), T, 0, stream>>>(f_bonds, nb, BFD, fb16, KBI);
  transpose_pad_kernel<<<cdiv((long long)HP * KBI, T), T, 0, stream>>>(W_i, BFD, KBI, wi_t);
  transpose_pad_kernel<<<cdiv((long long)HP * HP,  T), T, 0, stream>>>(W_h, HDIM, HP, wh_t);
  transpose_pad_kernel<<<cdiv((long long)HP * KAO, T), T, 0, stream>>>(W_o, AF + HDIM, KAO, wo_t);
  pad_bias_kernel<<<cdiv(HP, T), T, 0, stream>>>(b_o, biasp);

  // --- inp = f_bonds @ W_i ; msg = relu(inp) ---
  {
    dim3 g(HP / NBLK, cdiv(nb, MBLK));
    gemm_wmma_kernel<0><<<g, T, 0, stream>>>(fb16, KBI, nb, KBI / KSTEP, wi_t, KBI,
                                             nullptr, nullptr, inp, msg, nullptr, HP);
  }

  // --- message passing: DEPTH-1 = 2 iterations ---
  for (int d = 0; d < 2; ++d) {
    aggregate_kernel<<<cdiv((long long)na * (HP / 8), T), T, 0, stream>>>(msg, a2b, na, amsg);
    bond_build_kernel<<<cdiv((long long)nb * (HP / 8), T), T, 0, stream>>>(amsg, msg, b2a, b2revb, nb, bnd);
    dim3 g(HP / NBLK, cdiv(nb, MBLK));
    gemm_wmma_kernel<1><<<g, T, 0, stream>>>(bnd, HP, nb, HP / KSTEP, wh_t, HP,
                                             inp, nullptr, nullptr, msg, nullptr, HP);
  }

  // --- readout ---
  aggregate_kernel<<<cdiv((long long)na * (HP / 8), T), T, 0, stream>>>(msg, a2b, na, amsg);
  build_ainput_kernel<<<cdiv((long long)na * KAO, T), T, 0, stream>>>(f_atoms, amsg, na, ain);
  {
    dim3 g(HP / NBLK, cdiv(na, MBLK));
    gemm_wmma_kernel<2><<<g, T, 0, stream>>>(ain, KAO, na, KAO / KSTEP, wo_t, KAO,
                                             nullptr, biasp, nullptr, nullptr, hid, HP);
  }

  // --- per-molecule mean ---
  float* out = (float*)d_out;
  zero_f32_kernel<<<cdiv((long long)nm * HDIM, T), T, 0, stream>>>(out, nm * HDIM);
  zero_f32_kernel<<<cdiv(nm, T), T, 0, stream>>>(counts, nm);
  count_kernel<<<cdiv(na, T), T, 0, stream>>>(atom_mol, na, counts);
  scatter_hidden_kernel<<<cdiv((long long)na * HDIM, T), T, 0, stream>>>(hid, atom_mol, na, out);
  divide_kernel<<<cdiv((long long)nm * HDIM, T), T, 0, stream>>>(out, counts, nm);
}
